// MoRLlamaDecoderLayer_63445256896792
// MI455X (gfx1250) — matrix operand order repaired and checked by reading, not compiled, not run on hardware
//
#include <hip/hip_runtime.h>
#include <hip/hip_bf16.h>
#include <cstdint>

#define BB   2
#define SS   1024
#define DD   2048
#define HH   16
#define HDD  128
#define FFF  8192
#define RR   3
#define NEGV (-1.0e9f)

typedef __attribute__((ext_vector_type(16))) __bf16 v16bf;
typedef __attribute__((ext_vector_type(8)))  float  v8f;

#if __has_builtin(__builtin_amdgcn_global_load_async_to_lds_b128)
#define HAS_ASYNC_LDS 1
typedef int v4i_vs __attribute__((vector_size(16)));
typedef __attribute__((address_space(1))) v4i_vs*          as1_v4i;
typedef __attribute__((address_space(3))) v4i_vs*          as3_v4i;
typedef __attribute__((address_space(3))) unsigned short*  as3_u16;
#else
#define HAS_ASYNC_LDS 0
#endif

union BFrag {
  v16bf v;
  uint4 q[2];
  unsigned short u[16];
};

__device__ __forceinline__ unsigned short f2bf(float f){
  unsigned int uu = __float_as_uint(f);
  unsigned int rr = uu + 0x7FFFu + ((uu >> 16) & 1u);   // round-to-nearest-even bf16
  return (unsigned short)(rr >> 16);
}
__device__ __forceinline__ float bf2f(unsigned short h){
  return __uint_as_float(((unsigned int)h) << 16);
}
__device__ __forceinline__ v8f wmma_bf16(v16bf a, v16bf b, v8f c){
  // (neg_a, A, neg_b, B, c_mod, C, reuse_a, reuse_b)
  return __builtin_amdgcn_wmma_f32_16x16x32_bf16(false, a, false, b, (short)0, c, false, false);
}
__device__ __forceinline__ void wait_asynccnt0(){
#if HAS_ASYNC_LDS
#if __has_builtin(__builtin_amdgcn_s_wait_asynccnt)
  __builtin_amdgcn_s_wait_asynccnt(0);
#else
  asm volatile("s_wait_asynccnt 0" ::: "memory");
#endif
#endif
}

// ---------------------------------------------------------------------------
// GEMM: C[M,N] = A(bf16,[M,K]) @ B(f32,[K,N]) with fused epilogues.
// EPI 0: C bf16 = acc
// EPI 1: C f32  = acc + extra_f32   (residual add)
// EPI 2: C bf16 = silu(extra_bf16) * acc   (SwiGLU gate)
// A tile staged via GLOBAL_LOAD_ASYNC_TO_LDS_B128 (ASYNCcnt) when available.
// ---------------------------------------------------------------------------
template<int EPI>
__global__ __launch_bounds__(256)
void gemm_kernel(const unsigned short* __restrict__ A,
                 const float* __restrict__ Bw,
                 void* __restrict__ Cout,
                 const void* __restrict__ extra,
                 int M, int N, int K)
{
  __shared__ __align__(16) unsigned short ldsA[128 * 32];   // [row][k]
  __shared__ __align__(16) unsigned short ldsBT[128 * 32];  // [col][k] (transposed)
  const int tid   = threadIdx.x;
  const int lane  = tid & 31;
  const int wave  = tid >> 5;
  const int lhalf = lane >> 4;
  const int l15   = lane & 15;
  const int wr    = wave >> 1;        // 0..3  (M direction)
  const int wc    = wave & 1;         // 0..1  (N direction)
  const int mBlk  = blockIdx.y * 128;
  const int nBlk  = blockIdx.x * 128;

  v8f acc[2][4];
#pragma unroll
  for (int i = 0; i < 2; ++i)
#pragma unroll
    for (int j = 0; j < 4; ++j)
#pragma unroll
      for (int e = 0; e < 8; ++e) acc[i][j][e] = 0.0f;

  const int arow = tid >> 1;
  const int acol = (tid & 1) * 16;
#if HAS_ASYNC_LDS
  as3_u16 ldsA3 = (as3_u16)ldsA;    // addrspacecast generic -> LDS
#endif

  for (int k0 = 0; k0 < K; k0 += 32) {
    // Stage A tile: 128x32 bf16 (two b128 per thread)
    {
      const unsigned short* srcp = A + (size_t)(mBlk + arow) * K + k0 + acol;
#if HAS_ASYNC_LDS
      __builtin_amdgcn_global_load_async_to_lds_b128(
          (as1_v4i)(size_t)srcp,       (as3_v4i)(ldsA3 + arow * 32 + acol),     0, 0);
      __builtin_amdgcn_global_load_async_to_lds_b128(
          (as1_v4i)(size_t)(srcp + 8), (as3_v4i)(ldsA3 + arow * 32 + acol + 8), 0, 0);
#else
      const uint4* src = reinterpret_cast<const uint4*>(srcp);
      uint4* dst = reinterpret_cast<uint4*>(ldsA + arow * 32 + acol);
      uint4 x0 = src[0];
      uint4 x1 = src[1];
      dst[0] = x0; dst[1] = x1;
#endif
    }
    // Stage B tile: 32x128 f32 -> bf16, transposed into ldsBT[col][k]
#pragma unroll
    for (int i = 0; i < 4; ++i) {
      int f  = i * 256 + tid;
      int kr = f >> 5;          // 0..31
      int c4 = f & 31;          // 0..31 (float4 column)
      const float4 vv = *reinterpret_cast<const float4*>(Bw + (size_t)(k0 + kr) * N + nBlk + c4 * 4);
      ldsBT[(c4 * 4 + 0) * 32 + kr] = f2bf(vv.x);
      ldsBT[(c4 * 4 + 1) * 32 + kr] = f2bf(vv.y);
      ldsBT[(c4 * 4 + 2) * 32 + kr] = f2bf(vv.z);
      ldsBT[(c4 * 4 + 3) * 32 + kr] = f2bf(vv.w);
    }
    if (k0 + 32 < K)  // pull next weight tile toward L2 (global_prefetch_b8)
      __builtin_prefetch(Bw + (size_t)(k0 + 32 + (tid >> 3)) * N + nBlk + (tid & 7) * 16, 0, 1);
    wait_asynccnt0();
    __syncthreads();

    // A fragments: lane = M row, elems 0-7 -> K = khalf8+j, 8-15 -> K = 16+khalf8+j
    BFrag af[2];
#pragma unroll
    for (int i = 0; i < 2; ++i) {
      const unsigned short* p = ldsA + (wr * 32 + i * 16 + l15) * 32 + lhalf * 8;
      af[i].q[0] = *reinterpret_cast<const uint4*>(p);
      af[i].q[1] = *reinterpret_cast<const uint4*>(p + 16);
    }
    // B fragments: lane = N col, elems 0-7 -> K = khalf8+j, 8-15 -> K = 16+khalf8+j
    BFrag bfr[4];
#pragma unroll
    for (int j = 0; j < 4; ++j) {
      const unsigned short* p = ldsBT + (wc * 64 + j * 16 + l15) * 32 + lhalf * 8;
      bfr[j].q[0] = *reinterpret_cast<const uint4*>(p);
      bfr[j].q[1] = *reinterpret_cast<const uint4*>(p + 16);
    }
#pragma unroll
    for (int i = 0; i < 2; ++i)
#pragma unroll
      for (int j = 0; j < 4; ++j)
        acc[i][j] = wmma_bf16(af[i].v, bfr[j].v, acc[i][j]);
    __syncthreads();
  }

  // Epilogue. C layout: vgpr e -> M = e (lanes 0-15) / e+8 (lanes 16-31), N = l15.
#pragma unroll
  for (int i = 0; i < 2; ++i) {
#pragma unroll
    for (int j = 0; j < 4; ++j) {
      const int c = nBlk + wc * 64 + j * 16 + l15;
#pragma unroll
      for (int e = 0; e < 8; ++e) {
        const int r = mBlk + wr * 32 + i * 16 + lhalf * 8 + e;
        const size_t idx = (size_t)r * N + c;
        const float val = acc[i][j][e];
        if (EPI == 0) {
          ((unsigned short*)Cout)[idx] = f2bf(val);
        } else if (EPI == 1) {
          ((float*)Cout)[idx] = val + ((const float*)extra)[idx];
        } else {
          const float g = bf2f(((const unsigned short*)extra)[idx]);
          const float s = g / (1.0f + __expf(-g));
          ((unsigned short*)Cout)[idx] = f2bf(s * val);
        }
      }
    }
  }
}

// ---------------------------------------------------------------------------
// Flash attention, WMMA for QK^T and P*V. One wave = 16 query rows, 8 waves/WG.
// ---------------------------------------------------------------------------
__global__ __launch_bounds__(256)
void attn_kernel(const unsigned short* __restrict__ qb,
                 const unsigned short* __restrict__ kb,
                 const unsigned short* __restrict__ vb,
                 unsigned short* __restrict__ ob,
                 const int* __restrict__ top, int rd)
{
  __shared__ __align__(16) unsigned short ldsP[8][16 * 32];  // per-wave P scratch
  const int b     = blockIdx.z;
  const int h     = blockIdx.y;
  const int tid   = threadIdx.x;
  const int wave  = tid >> 5;
  const int lane  = tid & 31;
  const int lhalf = lane >> 4;
  const int l15   = lane & 15;
  const int q0    = blockIdx.x * 128 + wave * 16;

  // Q fragments (A layout), 4 chunks of K-dim 32 covering HD=128
  BFrag qf[4];
  const unsigned short* qbase = qb + ((size_t)(b * SS + q0 + l15)) * DD + h * HDD;
#pragma unroll
  for (int c = 0; c < 4; ++c) {
    const unsigned short* p = qbase + c * 32 + lhalf * 8;
    qf[c].q[0] = *reinterpret_cast<const uint4*>(p);
    qf[c].q[1] = *reinterpret_cast<const uint4*>(p + 16);
  }

  float m_run[8], l_run[8];
  v8f oacc[8];
#pragma unroll
  for (int e = 0; e < 8; ++e) { m_run[e] = -3.0e38f; l_run[e] = 0.0f; }
#pragma unroll
  for (int c = 0; c < 8; ++c)
#pragma unroll
    for (int e = 0; e < 8; ++e) oacc[c][e] = 0.0f;

  const int kEnd = q0 + 16;  // causal bound for this wave's rows
  for (int kb0 = 0; kb0 < kEnd; kb0 += 32) {
    v8f sc[2];
#pragma unroll
    for (int tk = 0; tk < 2; ++tk)
#pragma unroll
      for (int e = 0; e < 8; ++e) sc[tk][e] = 0.0f;

    // scores = Q @ K^T : B fragment lane = key col, elems 0-7 -> HD khalf8+j, 8-15 -> HD 16+khalf8+j
#pragma unroll
    for (int tk = 0; tk < 2; ++tk) {
      const unsigned short* kbase = kb + ((size_t)(b * SS + kb0 + tk * 16 + l15)) * DD + h * HDD;
#pragma unroll
      for (int c = 0; c < 4; ++c) {
        BFrag kf;
        const unsigned short* p = kbase + c * 32 + lhalf * 8;
        kf.q[0] = *reinterpret_cast<const uint4*>(p);
        kf.q[1] = *reinterpret_cast<const uint4*>(p + 16);
        sc[tk] = wmma_bf16(qf[c].v, kf.v, sc[tk]);
      }
    }

    // mask (causal + MoR selection) and scale
    const int key0 = kb0 + l15;
    const int key1 = kb0 + 16 + l15;
    const int sel0 = top[b * SS + key0] >= rd;
    const int sel1 = top[b * SS + key1] >= rd;
#pragma unroll
    for (int e = 0; e < 8; ++e) {
      const int qr = q0 + lhalf * 8 + e;
      const float s0 = sc[0][e] * 0.08838834764831845f;  // 1/sqrt(128)
      const float s1 = sc[1][e] * 0.08838834764831845f;
      sc[0][e] = (sel0 && key0 <= qr) ? s0 : NEGV;
      sc[1][e] = (sel1 && key1 <= qr) ? s1 : NEGV;
    }

    // online softmax (row = (vgpr e, lane half); reduce across the 16 lanes)
#pragma unroll
    for (int e = 0; e < 8; ++e) {
      float bm = fmaxf(sc[0][e], sc[1][e]);
#pragma unroll
      for (int off = 1; off < 16; off <<= 1)
        bm = fmaxf(bm, __shfl_xor(bm, off, 32));
      const float mnew = fmaxf(m_run[e], bm);
      const float corr = __expf(m_run[e] - mnew);
      const float p0 = __expf(sc[0][e] - mnew);
      const float p1 = __expf(sc[1][e] - mnew);
      float ps = p0 + p1;
#pragma unroll
      for (int off = 1; off < 16; off <<= 1)
        ps += __shfl_xor(ps, off, 32);
      l_run[e] = l_run[e] * corr + ps;
      m_run[e] = mnew;
#pragma unroll
      for (int c = 0; c < 8; ++c) oacc[c][e] *= corr;
      // write P to LDS in [row][key] so it can be re-read in A-fragment layout
      ldsP[wave][(lhalf * 8 + e) * 32 + l15]      = f2bf(p0);
      ldsP[wave][(lhalf * 8 + e) * 32 + 16 + l15] = f2bf(p1);
    }
    asm volatile("s_wait_dscnt 0" ::: "memory");  // wave-local LDS transpose fence

    BFrag pf;
    {
      const unsigned short* p = &ldsP[wave][l15 * 32 + lhalf * 16];
      pf.q[0] = *reinterpret_cast<const uint4*>(p);
      pf.q[1] = *reinterpret_cast<const uint4*>(p + 8);
    }
    // oacc += P @ V : B fragment lane = HD col, elems = 16 keys (gather)
#pragma unroll
    for (int c = 0; c < 8; ++c) {
      BFrag vf;
#pragma unroll
      for (int j = 0; j < 16; ++j)
        vf.u[j] = vb[((size_t)(b * SS + kb0 + lhalf * 16 + j)) * DD + h * HDD + c * 16 + l15];
      oacc[c] = wmma_bf16(pf.v, vf.v, oacc[c]);
    }
  }

  // normalize and store (bf16)
#pragma unroll
  for (int c = 0; c < 8; ++c) {
#pragma unroll
    for (int e = 0; e < 8; ++e) {
      const int qr = q0 + lhalf * 8 + e;
      const float o = oacc[c][e] / l_run[e];
      ob[((size_t)(b * SS + qr)) * DD + h * HDD + c * 16 + l15] = f2bf(o);
    }
  }
}

// ---------------------------------------------------------------------------
// Elementwise / reduction kernels
// ---------------------------------------------------------------------------
__global__ __launch_bounds__(256)
void router_kernel(const float* __restrict__ x, const float* __restrict__ wr,
                   int* __restrict__ top, float* __restrict__ wgt)
{
  const int t = blockIdx.x;
  const int tid = threadIdx.x;
  const float* xr = x + (size_t)t * DD;
  float s0 = 0.f, s1 = 0.f, s2 = 0.f;
  for (int d = tid; d < DD; d += 256) {
    const float v = xr[d];                 // TEMP == 1
    s0 += v * wr[d * 3 + 0];
    s1 += v * wr[d * 3 + 1];
    s2 += v * wr[d * 3 + 2];
  }
  __shared__ float red[3 * 256];
  red[tid] = s0; red[256 + tid] = s1; red[512 + tid] = s2;
  __syncthreads();
  for (int st = 128; st > 0; st >>= 1) {
    if (tid < st) {
      red[tid]       += red[tid + st];
      red[256 + tid] += red[256 + tid + st];
      red[512 + tid] += red[512 + tid + st];
    }
    __syncthreads();
  }
  if (tid == 0) {
    const float l0 = red[0], l1 = red[256], l2 = red[512];
    const float mx = fmaxf(l0, fmaxf(l1, l2));
    const float e0 = __expf(l0 - mx), e1 = __expf(l1 - mx), e2 = __expf(l2 - mx);
    const float den = e0 + e1 + e2;
    int idx = 0; float best = e0;
    if (e1 > best) { best = e1; idx = 1; }
    if (e2 > best) { best = e2; idx = 2; }
    top[t] = idx;
    wgt[t] = best / den;                   // ALPHA == 1
  }
}

__global__ __launch_bounds__(256)
void rmsnorm_kernel(const float* __restrict__ x, const float* __restrict__ w,
                    unsigned short* __restrict__ out)
{
  const int t = blockIdx.x;
  const int tid = threadIdx.x;
  const float* xr = x + (size_t)t * DD;
  float vals[8];
  float ss = 0.f;
#pragma unroll
  for (int i = 0; i < 8; ++i) {
    const float v = xr[tid + i * 256];
    vals[i] = v;
    ss += v * v;
  }
  __shared__ float red[256];
  red[tid] = ss;
  __syncthreads();
  for (int st = 128; st > 0; st >>= 1) {
    if (tid < st) red[tid] += red[tid + st];
    __syncthreads();
  }
  const float scale = rsqrtf(red[0] * (1.0f / DD) + 1e-6f);
#pragma unroll
  for (int i = 0; i < 8; ++i) {
    const int d = tid + i * 256;
    out[(size_t)t * DD + d] = f2bf(vals[i] * scale * w[d]);
  }
}

__global__ __launch_bounds__(256)
void rope_kernel(unsigned short* __restrict__ q, unsigned short* __restrict__ k,
                 const float* __restrict__ cosb, const float* __restrict__ sinb)
{
  const int i  = blockIdx.x * 256 + threadIdx.x;  // over T*H*64
  const int dd = i & 63;
  const int h  = (i >> 6) & (HH - 1);
  const int t  = i >> 10;
  const int s  = t & (SS - 1);
  const float c1 = cosb[s * HDD + dd],      sn1 = sinb[s * HDD + dd];
  const float c2 = cosb[s * HDD + 64 + dd], sn2 = sinb[s * HDD + 64 + dd];
  const size_t base = (size_t)t * DD + h * HDD + dd;
  const float q1 = bf2f(q[base]), q2 = bf2f(q[base + 64]);
  q[base]      = f2bf(q1 * c1 - q2 * sn1);
  q[base + 64] = f2bf(q2 * c2 + q1 * sn2);
  const float k1 = bf2f(k[base]), k2 = bf2f(k[base + 64]);
  k[base]      = f2bf(k1 * c1 - k2 * sn1);
  k[base + 64] = f2bf(k2 * c2 + k1 * sn2);
}

__global__ __launch_bounds__(256)
void init_kernel(const float* __restrict__ x, float* __restrict__ cur, float* __restrict__ fin)
{
  const size_t i = (size_t)blockIdx.x * 256 + threadIdx.x;
  const float v = x[i];
  cur[i] = v;
  fin[i] = v;
}

__global__ __launch_bounds__(256)
void merge_kernel(const float* __restrict__ outb, float* __restrict__ cur,
                  float* __restrict__ fin, const int* __restrict__ top,
                  const float* __restrict__ wgt, int rd)
{
  const size_t i = (size_t)blockIdx.x * 256 + threadIdx.x;
  const int t = (int)(i >> 11);                  // D = 2048
  const int tp = top[t];
  const float o = outb[i];
  if (tp == rd) fin[i] = o * wgt[t];
  if (tp >= rd) cur[i] = o;                      // cur = where(sel, out, cur)
}

// ---------------------------------------------------------------------------
extern "C" void kernel_launch(void* const* d_in, const int* in_sizes, int n_in,
                              void* d_out, int out_size, void* d_ws, size_t ws_size,
                              hipStream_t stream)
{
  const float* x    = (const float*)d_in[0];
  const float* cosb = (const float*)d_in[1];
  const float* sinb = (const float*)d_in[2];
  const float* wrt  = (const float*)d_in[3];
  const float* ln1  = (const float*)d_in[4];
  const float* ln2  = (const float*)d_in[5];
  const float* Wq   = (const float*)d_in[6];
  const float* Wk   = (const float*)d_in[7];
  const float* Wv   = (const float*)d_in[8];
  const float* Wo   = (const float*)d_in[9];
  const float* Wg   = (const float*)d_in[10];
  const float* Wu   = (const float*)d_in[11];
  const float* Wd   = (const float*)d_in[12];
  float* finalx = (float*)d_out;
  (void)in_sizes; (void)n_in; (void)out_size; (void)ws_size;

  char* ws = (char*)d_ws;
  size_t off = 0;
  auto take = [&](size_t bytes) -> void* {
    void* p = ws + off;
    off += (bytes + 255) & ~(size_t)255;
    return p;
  };
  const size_t T = (size_t)BB * SS;  // 2048 tokens
  float*          cur      = (float*)take(T * DD * 4);
  float*          xm       = (float*)take(T * DD * 4);
  float*          outb     = (float*)take(T * DD * 4);
  unsigned short* hb       = (unsigned short*)take(T * DD * 2);
  unsigned short* qbuf     = (unsigned short*)take(T * DD * 2);
  unsigned short* kbuf     = (unsigned short*)take(T * DD * 2);
  unsigned short* vbuf     = (unsigned short*)take(T * DD * 2);
  unsigned short* obuf     = (unsigned short*)take(T * DD * 2);
  unsigned short* gbuf     = (unsigned short*)take(T * (size_t)FFF * 2);
  unsigned short* gatedbuf = (unsigned short*)take(T * (size_t)FFF * 2);
  int*            topb     = (int*)take(T * 4);
  float*          wgtb     = (float*)take(T * 4);

  const dim3 blk(256);
  init_kernel<<<dim3((unsigned)(T * DD / 256)), blk, 0, stream>>>(x, cur, finalx);
  router_kernel<<<dim3((unsigned)T), blk, 0, stream>>>(x, wrt, topb, wgtb);

  for (int r = 0; r < RR; ++r) {
    rmsnorm_kernel<<<dim3((unsigned)T), blk, 0, stream>>>(cur, ln1, hb);
    gemm_kernel<0><<<dim3(DD / 128, (unsigned)(T / 128)), blk, 0, stream>>>(hb, Wq, qbuf, nullptr, (int)T, DD, DD);
    gemm_kernel<0><<<dim3(DD / 128, (unsigned)(T / 128)), blk, 0, stream>>>(hb, Wk, kbuf, nullptr, (int)T, DD, DD);
    gemm_kernel<0><<<dim3(DD / 128, (unsigned)(T / 128)), blk, 0, stream>>>(hb, Wv, vbuf, nullptr, (int)T, DD, DD);
    rope_kernel<<<dim3((unsigned)(T * HH * 64 / 256)), blk, 0, stream>>>(qbuf, kbuf, cosb, sinb);
    attn_kernel<<<dim3(SS / 128, HH, BB), blk, 0, stream>>>(qbuf, kbuf, vbuf, obuf, topb, r);
    gemm_kernel<1><<<dim3(DD / 128, (unsigned)(T / 128)), blk, 0, stream>>>(obuf, Wo, xm, cur, (int)T, DD, DD);
    rmsnorm_kernel<<<dim3((unsigned)T), blk, 0, stream>>>(xm, ln2, hb);
    gemm_kernel<0><<<dim3(FFF / 128, (unsigned)(T / 128)), blk, 0, stream>>>(hb, Wg, gbuf, nullptr, (int)T, FFF, DD);
    gemm_kernel<2><<<dim3(FFF / 128, (unsigned)(T / 128)), blk, 0, stream>>>(hb, Wu, gatedbuf, gbuf, (int)T, FFF, DD);
    gemm_kernel<1><<<dim3(DD / 128, (unsigned)(T / 128)), blk, 0, stream>>>(gatedbuf, Wd, outb, xm, (int)T, DD, FFF);
    merge_kernel<<<dim3((unsigned)(T * DD / 256)), blk, 0, stream>>>(outb, cur, finalx, topb, wgtb, r);
  }
}
